// MultiheadAttentionLayer_87462714015841
// MI455X (gfx1250) — matrix operand order repaired and checked by reading, not compiled, not run on hardware
//
#include <hip/hip_runtime.h>
#include <hip/hip_bf16.h>

typedef __attribute__((ext_vector_type(16))) _Float16 v16h;
typedef __attribute__((ext_vector_type(8)))  _Float16 v8h;
typedef __attribute__((ext_vector_type(8)))  float    v8f;
typedef __attribute__((ext_vector_type(4)))  float    v4f;
typedef __attribute__((ext_vector_type(4)))  int      v4i;

union V16 { v16h v; v8h h[2]; };

// ---- optional gfx1250 async global->LDS copy path (ASYNCcnt-tracked) ----
#if defined(__has_builtin)
#  if __has_builtin(__builtin_amdgcn_global_load_async_to_lds_b128)
#    define HAVE_ASYNC_COPY 1
#  endif
#endif
#ifndef HAVE_ASYNC_COPY
#  define HAVE_ASYNC_COPY 0
#endif

#if HAVE_ASYNC_COPY
// signature (from hipcc diagnostic): (v4i addrspace(1)*, v4i addrspace(3)*, Imm, Imm)
typedef __attribute__((address_space(1))) v4i* gv4i_p;
typedef __attribute__((address_space(3))) v4i* lv4i_p;
#  define AS_G(p) ((gv4i_p)(void*)(p))
#  define AS_L(p) ((lv4i_p)(void*)(p))
#  if __has_builtin(__builtin_amdgcn_s_wait_asynccnt)
#    define WAIT_ASYNC() __builtin_amdgcn_s_wait_asynccnt(0)
#  else
#    define WAIT_ASYNC() asm volatile("s_wait_asynccnt 0x0" ::: "memory")
#  endif
#endif

// A-operand (16x32 f16), per ISA table: lane-lo K {0..7,16..23}, lane-hi K {8..15,24..31}
__device__ __forceinline__ v16h lds_a_operand(const _Float16* rowbase, int c0, int hi) {
  V16 u;
  u.h[0] = *(const v8h*)(rowbase + c0 + hi * 8);
  u.h[1] = *(const v8h*)(rowbase + c0 + 16 + hi * 8);
  return u.v;
}
// B-operand (32x16 f16): lane-lo col n K {0..7,16..23}, lane-hi same col K {8..15,24..31}
__device__ __forceinline__ v16h lds_b_operand(const _Float16* colbase, int c0, int hi) {
  const _Float16* p = colbase + c0 + hi * 8;
  V16 u;
  u.h[0] = *(const v8h*)(p);
  u.h[1] = *(const v8h*)(p + 16);
  return u.v;
}
#define WMMA_F32_F16(a, b, c) \
  __builtin_amdgcn_wmma_f32_16x16x32_f16(false, (a), false, (b), (short)0, (c), false, false)

#define BB 4
#define TQ 2048
#define TKK 2048
#define DM 512
#define HH 8
#define DH 64

// ---------------------------------------------------------------------------
// Kernel 1: C = f16(relu(A @ W + bias)), A [M,512] fp32, W [512,512] fp32.
// 128x128 tile per block, K-step 32, 8 waves (4m x 2n), each wave 32x64 out.
// (fp32->f16 conversion required, so staging stays on the VALU path)
// ---------------------------------------------------------------------------
__global__ __launch_bounds__(256) void proj_gemm_relu_f16(
    const float* __restrict__ A, const float* __restrict__ W,
    const float* __restrict__ bias, _Float16* __restrict__ out) {
  __shared__ __attribute__((aligned(32))) _Float16 As[128 * 32];
  __shared__ __attribute__((aligned(32))) _Float16 Wt[128 * 32];  // [n][k]
  const int tid = threadIdx.x;
  const int lane = tid & 31, w = tid >> 5;
  const int lh = lane & 15, hi = lane >> 4;
  const int m0 = blockIdx.y * 128, n0 = blockIdx.x * 128;
  const int wm = (w >> 1) * 32, wn = (w & 1) * 64;

  const v8f vzero = {0.f, 0.f, 0.f, 0.f, 0.f, 0.f, 0.f, 0.f};
  v8f acc[2][4];
  for (int mi = 0; mi < 2; mi++)
    for (int ni = 0; ni < 4; ni++) acc[mi][ni] = vzero;

  for (int k0 = 0; k0 < DM; k0 += 32) {
    __syncthreads();
    // A tile 128x32 fp32 -> f16 LDS (row-major)
    for (int i = 0; i < 4; i++) {
      int e = tid * 16 + i * 4;
      int r = e >> 5, c = e & 31;
      v4f f = *(const v4f*)&A[(size_t)(m0 + r) * DM + k0 + c];
      As[r * 32 + c + 0] = (_Float16)f[0];
      As[r * 32 + c + 1] = (_Float16)f[1];
      As[r * 32 + c + 2] = (_Float16)f[2];
      As[r * 32 + c + 3] = (_Float16)f[3];
    }
    // W tile 32x128 fp32 -> transposed f16 LDS Wt[n][k]
    for (int i = 0; i < 4; i++) {
      int e = tid * 16 + i * 4;
      int k = e >> 7, n = e & 127;
      v4f f = *(const v4f*)&W[(size_t)(k0 + k) * DM + n0 + n];
      Wt[(n + 0) * 32 + k] = (_Float16)f[0];
      Wt[(n + 1) * 32 + k] = (_Float16)f[1];
      Wt[(n + 2) * 32 + k] = (_Float16)f[2];
      Wt[(n + 3) * 32 + k] = (_Float16)f[3];
    }
    __syncthreads();
    v16h a[2], bb[4];
    for (int mi = 0; mi < 2; mi++)
      a[mi] = lds_a_operand(&As[(wm + mi * 16 + lh) * 32], 0, hi);
    for (int ni = 0; ni < 4; ni++)
      bb[ni] = lds_b_operand(&Wt[(wn + ni * 16 + lh) * 32], 0, hi);
    for (int mi = 0; mi < 2; mi++)
      for (int ni = 0; ni < 4; ni++)
        acc[mi][ni] = WMMA_F32_F16(a[mi], bb[ni], acc[mi][ni]);
  }
  // epilogue: + bias, relu, f16 store
  for (int mi = 0; mi < 2; mi++)
    for (int ni = 0; ni < 4; ni++) {
      int n = n0 + wn + ni * 16 + lh;
      float bv = bias[n];
      for (int r = 0; r < 8; r++) {
        int row = m0 + wm + mi * 16 + r + 8 * hi;
        float v = acc[mi][ni][r] + bv;
        v = v > 0.f ? v : 0.f;
        out[(size_t)row * DM + n] = (_Float16)v;
      }
    }
}

// ---------------------------------------------------------------------------
// Kernel 2: flash attention per (b, h, 64-query tile).
// 8 waves = 4 q-tiles x 2 column-halves. Key chunks of 64, online softmax.
// Q/K tiles (already f16) staged with async global->LDS copies when available.
// ---------------------------------------------------------------------------
__global__ __launch_bounds__(256) void mha_flash(
    const _Float16* __restrict__ Qf, const _Float16* __restrict__ Kf,
    const _Float16* __restrict__ Vf, float* __restrict__ Ows) {
  __shared__ __attribute__((aligned(32))) char smem[41 * 1024];
  _Float16* Ks = (_Float16*)(smem);              // 64x64 [j][d]
  _Float16* Vt = (_Float16*)(smem + 8 * 1024);   // 64x64 [d][j]
  float* Sb = (float*)(smem + 16 * 1024);        // 64x64 f32 (Q staging at start)
  _Float16* Pb = (_Float16*)(smem + 32 * 1024);  // 64x64 f16
  float* kbias = (float*)(smem + 40 * 1024);            // 64
  float* rowscale = (float*)(smem + 40 * 1024 + 256);   // 64
  float* qmaskv = (float*)(smem + 40 * 1024 + 512);     // 64
  float* rowl = (float*)(smem + 40 * 1024 + 768);       // 64

  const int tid = threadIdx.x;
  const int lane = tid & 31, w = tid >> 5;
  const int lh = lane & 15, hi = lane >> 4;
  const int qi = w >> 1, half = w & 1;
  const int qb = blockIdx.x * 64;
  const int h = blockIdx.y;
  const int b = blockIdx.z;

  // stage Q chunk (64x64 f16) into Sb region
  _Float16* Qs = (_Float16*)Sb;
#if HAVE_ASYNC_COPY
  for (int i = 0; i < 2; i++) {
    int e = tid * 16 + i * 8;
    int r = e >> 6, c = e & 63;
    __builtin_amdgcn_global_load_async_to_lds_b128(
        AS_G(&Qf[((size_t)(b * TQ + qb + r)) * DM + h * DH + c]),
        AS_L(&Qs[r * 64 + c]), 0, 0);
  }
  WAIT_ASYNC();
#else
  for (int i = 0; i < 2; i++) {
    int e = tid * 16 + i * 8;
    int r = e >> 6, c = e & 63;
    *(v8h*)&Qs[r * 64 + c] =
        *(const v8h*)&Qf[((size_t)(b * TQ + qb + r)) * DM + h * DH + c];
  }
#endif
  __syncthreads();
  v16h qa[2];
  qa[0] = lds_a_operand(&Qs[(qi * 16 + lh) * 64], 0, hi);
  qa[1] = lds_a_operand(&Qs[(qi * 16 + lh) * 64], 32, hi);
  if (tid < 64) {
    float s = 0.f;
    for (int j = 0; j < 64; j++) s += (float)Qs[tid * 64 + j];
    qmaskv[tid] = s > 0.f ? 1.f : 0.f;  // relu'd Q: sum == sum|Q|
  }
  __syncthreads();

  const v8f vzero = {0.f, 0.f, 0.f, 0.f, 0.f, 0.f, 0.f, 0.f};
  v8f o[2] = {vzero, vzero};
  float m_run = -1e30f, l_run = 0.f;

  for (int kc = 0; kc < TKK; kc += 64) {
    // stage K (row-major, async if available) and V (transposed, VALU)
#if HAVE_ASYNC_COPY
    for (int i = 0; i < 2; i++) {
      int e = tid * 16 + i * 8;
      int r = e >> 6, c = e & 63;
      __builtin_amdgcn_global_load_async_to_lds_b128(
          AS_G(&Kf[((size_t)(b * TKK + kc + r)) * DM + h * DH + c]),
          AS_L(&Ks[r * 64 + c]), 0, 0);
    }
    for (int i = 0; i < 2; i++) {  // V transpose overlaps in-flight K copy
      int e = tid * 16 + i * 8;
      int r = e >> 6, c = e & 63;
      v8h vv = *(const v8h*)&Vf[((size_t)(b * TKK + kc + r)) * DM + h * DH + c];
      for (int q = 0; q < 8; q++) Vt[(c + q) * 64 + r] = vv[q];
    }
    WAIT_ASYNC();
#else
    for (int i = 0; i < 2; i++) {
      int e = tid * 16 + i * 8;
      int r = e >> 6, c = e & 63;
      *(v8h*)&Ks[r * 64 + c] =
          *(const v8h*)&Kf[((size_t)(b * TKK + kc + r)) * DM + h * DH + c];
      v8h vv = *(const v8h*)&Vf[((size_t)(b * TKK + kc + r)) * DM + h * DH + c];
      for (int q = 0; q < 8; q++) Vt[(c + q) * 64 + r] = vv[q];
    }
#endif
    __syncthreads();
    if (tid < 64) {  // key mask bias from relu'd K row sums
      float s = 0.f;
      for (int j = 0; j < 64; j++) s += (float)Ks[tid * 64 + j];
      kbias[tid] = s > 0.f ? 0.f : -1e9f;
    }
    __syncthreads();
    // S = Q @ K^T * (1/sqrt(64)) + kbias
    for (int jt = 0; jt < 2; jt++) {
      int jn = half * 2 + jt;
      v8f s = vzero;
      for (int k0 = 0; k0 < 2; k0++)
        s = WMMA_F32_F16(qa[k0], lds_b_operand(&Ks[(jn * 16 + lh) * 64], k0 * 32, hi), s);
      float kb = kbias[jn * 16 + lh];
      for (int r = 0; r < 8; r++)
        Sb[(qi * 16 + r + 8 * hi) * 64 + jn * 16 + lh] = s[r] * 0.125f + kb;
    }
    __syncthreads();
    // online softmax, one thread per query row
    if (tid < 64) {
      const float* sr = &Sb[tid * 64];
      float cm = -1e30f;
      for (int j = 0; j < 64; j++) cm = fmaxf(cm, sr[j]);
      float mn = fmaxf(m_run, cm);
      float corr = __expf(m_run - mn);
      float ssum = 0.f;
      for (int j = 0; j < 64; j++) {
        float p = __expf(sr[j] - mn);
        ssum += p;
        Pb[tid * 64 + j] = (_Float16)p;
      }
      l_run = l_run * corr + ssum;
      m_run = mn;
      rowscale[tid] = corr;
    }
    __syncthreads();
    // rescale O, accumulate O += P @ V
    float corr8[8];
    for (int r = 0; r < 8; r++) corr8[r] = rowscale[qi * 16 + r + 8 * hi];
    for (int d = 0; d < 2; d++)
      for (int r = 0; r < 8; r++) o[d][r] *= corr8[r];
    v16h pa[2];
    pa[0] = lds_a_operand(&Pb[(qi * 16 + lh) * 64], 0, hi);
    pa[1] = lds_a_operand(&Pb[(qi * 16 + lh) * 64], 32, hi);
    for (int d = 0; d < 2; d++) {
      int dn = half * 2 + d;
      for (int k0 = 0; k0 < 2; k0++)
        o[d] = WMMA_F32_F16(pa[k0], lds_b_operand(&Vt[(dn * 16 + lh) * 64], k0 * 32, hi), o[d]);
    }
    __syncthreads();
  }
  if (tid < 64) rowl[tid] = l_run;
  __syncthreads();
  for (int d = 0; d < 2; d++) {
    int dn = half * 2 + d;
    for (int r = 0; r < 8; r++) {
      int row = qi * 16 + r + 8 * hi;
      float val = o[d][r] * qmaskv[row] / rowl[row];  // rowl >= 1 always
      Ows[((size_t)(b * TQ) + qb + row) * DM + h * DH + dn * 16 + lh] = val;
    }
  }
}

// ---------------------------------------------------------------------------
// Kernel 3: out = LayerNorm(O + queries) over D=512, eps=1e-3
// ---------------------------------------------------------------------------
__global__ __launch_bounds__(256) void residual_layernorm(
    const float* __restrict__ Ows, const float* __restrict__ queries,
    const float* __restrict__ gamma, const float* __restrict__ beta,
    float* __restrict__ out) {
  __shared__ float r1[256];
  __shared__ float r2[256];
  const int t = threadIdx.x;
  const size_t row = blockIdx.x;
  float x0 = Ows[row * DM + t] + queries[row * DM + t];
  float x1 = Ows[row * DM + t + 256] + queries[row * DM + t + 256];
  r1[t] = x0 + x1;
  r2[t] = x0 * x0 + x1 * x1;
  __syncthreads();
  for (int s = 128; s > 0; s >>= 1) {
    if (t < s) { r1[t] += r1[t + s]; r2[t] += r2[t + s]; }
    __syncthreads();
  }
  float mu = r1[0] * (1.f / 512.f);
  float var = r2[0] * (1.f / 512.f) - mu * mu;
  float inv = rsqrtf(var + 1e-3f);
  out[row * DM + t] = (x0 - mu) * inv * gamma[t] + beta[t];
  out[row * DM + t + 256] = (x1 - mu) * inv * gamma[t + 256] + beta[t + 256];
}

extern "C" void kernel_launch(void* const* d_in, const int* in_sizes, int n_in,
                              void* d_out, int out_size, void* d_ws, size_t ws_size,
                              hipStream_t stream) {
  const float* queries = (const float*)d_in[0];
  const float* keys = (const float*)d_in[1];
  const float* Wq = (const float*)d_in[2];
  const float* bq = (const float*)d_in[3];
  const float* Wk = (const float*)d_in[4];
  const float* bk = (const float*)d_in[5];
  const float* Wv = (const float*)d_in[6];
  const float* bv = (const float*)d_in[7];
  const float* gamma = (const float*)d_in[8];
  const float* beta = (const float*)d_in[9];
  float* out = (float*)d_out;

  char* ws = (char*)d_ws;
  _Float16* Qf = (_Float16*)(ws);                        // 8 MB
  _Float16* Kf = (_Float16*)(ws + 8ull * 1024 * 1024);   // 8 MB
  _Float16* Vf = (_Float16*)(ws + 16ull * 1024 * 1024);  // 8 MB
  float* Ows = (float*)(ws + 24ull * 1024 * 1024);       // 16 MB

  dim3 blk(256);
  dim3 g1(DM / 128, (BB * TQ) / 128);  // (4, 64)
  proj_gemm_relu_f16<<<g1, blk, 0, stream>>>(queries, Wq, bq, Qf);
  proj_gemm_relu_f16<<<g1, blk, 0, stream>>>(keys, Wk, bk, Kf);
  proj_gemm_relu_f16<<<g1, blk, 0, stream>>>(keys, Wv, bv, Vf);
  mha_flash<<<dim3(TQ / 64, HH, BB), blk, 0, stream>>>(Qf, Kf, Vf, Ows);
  residual_layernorm<<<dim3(BB * TQ), blk, 0, stream>>>(Ows, queries, gamma, beta, out);
}
